// RQwenAttention_74715251081268
// MI455X (gfx1250) — matrix operand order repaired and checked by reading, not compiled, not run on hardware
//
#include <hip/hip_runtime.h>
#include <math.h>

// ---------------------------------------------------------------------------
// RQwenAttention for MI455X (gfx1250, wave32, WMMA f32_16x16x32_f16)
// ---------------------------------------------------------------------------

typedef _Float16 v16h __attribute__((ext_vector_type(16)));
typedef _Float16 v8h  __attribute__((ext_vector_type(8)));
typedef _Float16 v4h  __attribute__((ext_vector_type(4)));
typedef float    v8f  __attribute__((ext_vector_type(8)));

#define BB    2
#define SS    2048
#define EMBD  2048
#define HH    16
#define DDIM  128
#define NEXPT 8
#define RRNK  16
#define MTOK  (BB*SS)        // 4096 tokens
#define HD    (HH*DDIM)      // 2048
#define NLORA (NEXPT*RRNK)   // 128

__device__ __forceinline__ v8f wmma16(v16h a, v16h b, v8f c) {
  // D = A(16x32 f16) * B(32x16 f16) + C(16x16 f32)
  return __builtin_amdgcn_wmma_f32_16x16x32_f16(false, a, false, b,
                                                (short)0, c, false, false);
}

// A-operand 16x32 f16 fragment. Source stored [M][K] row-major (stride in f16).
// ISA layout: lane L -> row m = L%16, half = L/16; K chunks
// {half*8 .. half*8+7} and {16+half*8 .. 16+half*8+7}.
__device__ __forceinline__ v16h loadFragA(const _Float16* base, int stride,
                                          int li, int half) {
  const _Float16* p = base + (size_t)li * stride + half * 8;
  v8h lo = *(const v8h*)p;
  v8h hi = *(const v8h*)(p + 16);
  return __builtin_shufflevector(lo, hi, 0,1,2,3,4,5,6,7,8,9,10,11,12,13,14,15);
}

// B-operand 32x16 f16 fragment. Source stored [N][K] row-major (i.e. B^T).
// ISA layout: lane L -> col n = L%16, half = L/16; K chunks
// {half*8 .. half*8+7} and {16+half*8 .. 16+half*8+7}.
__device__ __forceinline__ v16h loadFragB(const _Float16* base, int stride,
                                          int li, int half) {
  const _Float16* p = base + (size_t)li * stride + half * 8;
  v8h lo = *(const v8h*)p;
  v8h hi = *(const v8h*)(p + 16);
  return __builtin_shufflevector(lo, hi, 0,1,2,3,4,5,6,7,8,9,10,11,12,13,14,15);
}

// ---------------------------------------------------------------------------
// f32 -> f16 convert (vectorized x4)
// ---------------------------------------------------------------------------
__global__ __launch_bounds__(256) void cvt_f32_to_f16(
    const float* __restrict__ in, _Float16* __restrict__ out, int n4) {
  int i = blockIdx.x * blockDim.x + threadIdx.x;
  if (i < n4) {
    float4 f = ((const float4*)in)[i];
    v4h h = {(_Float16)f.x, (_Float16)f.y, (_Float16)f.z, (_Float16)f.w};
    ((v4h*)out)[i] = h;
  }
}

// ---------------------------------------------------------------------------
// WMMA GEMM: C[M,N] (+)= scale * A[M,K]_f16 @ Bw[N,K]_f32^T (+ bias) (* gate)
// Tile: 128(M) x 64(N), 8 wave32s, K stepped by 32 through LDS.
// GATE: multiply each 16-wide column tile by gate[token, coltile] (R == 16).
// ---------------------------------------------------------------------------
#define MT 128
#define NT 64
#define KT 32
#define LDST 40   // padded LDS row stride in f16

template<bool OUT_F16, bool GATE, bool ACC>
__global__ __launch_bounds__(256) void gemm_kernel(
    const _Float16* __restrict__ A, const float* __restrict__ Bw,
    const float* __restrict__ bias, const float* __restrict__ gate,
    float scale, void* __restrict__ Cout, int M, int N, int K) {
  __shared__ _Float16 At[MT * LDST];
  __shared__ _Float16 Bt[NT * LDST];
  const int n0 = blockIdx.x * NT, m0 = blockIdx.y * MT;
  const int tid = threadIdx.x;
  const int wave = tid >> 5, lane = tid & 31;
  const int li = lane & 15, half = lane >> 4;

  v8f acc[4];
  v8f zero = {0.f, 0.f, 0.f, 0.f, 0.f, 0.f, 0.f, 0.f};
#pragma unroll
  for (int ct = 0; ct < 4; ct++) acc[ct] = zero;

  for (int k0 = 0; k0 < K; k0 += KT) {
    __syncthreads();
    // Stage A tile: 128 rows x 32 f16 (16B chunks)
#pragma unroll
    for (int it = 0; it < 2; it++) {
      int c = tid + it * 256;
      int row = c >> 2, cc = c & 3;
      const _Float16* src = A + (size_t)(m0 + row) * K + k0 + cc * 8;
      *(v8h*)(At + row * LDST + cc * 8) = *(const v8h*)src;
      if (k0 + KT < K) __builtin_prefetch(src + KT, 0, 0);
    }
    // Stage B tile: 64 rows (output cols) x 32, converting f32 -> f16
    {
      int row = tid >> 2, cc = tid & 3;
      const float* src = Bw + (size_t)(n0 + row) * K + k0 + cc * 8;
      float4 f0 = *(const float4*)src;
      float4 f1 = *(const float4*)(src + 4);
      v8h d = {(_Float16)f0.x, (_Float16)f0.y, (_Float16)f0.z, (_Float16)f0.w,
               (_Float16)f1.x, (_Float16)f1.y, (_Float16)f1.z, (_Float16)f1.w};
      *(v8h*)(Bt + row * LDST + cc * 8) = d;
      if (k0 + KT < K) __builtin_prefetch(src + KT, 0, 0);
    }
    __syncthreads();
    v16h af = loadFragA(At + (wave * 16) * LDST, LDST, li, half);
    v16h bf[4];
#pragma unroll
    for (int ct = 0; ct < 4; ct++)
      bf[ct] = loadFragB(Bt + (ct * 16) * LDST, LDST, li, half);
#pragma unroll
    for (int ct = 0; ct < 4; ct++)
      acc[ct] = wmma16(af, bf[ct], acc[ct]);
  }

  // Epilogue. C layout: lane L holds col n = L%16; vgpr r -> row (L/16)*8 + r.
#pragma unroll
  for (int ct = 0; ct < 4; ct++) {
    int n = n0 + ct * 16 + li;
    float bval = (!ACC && bias != nullptr) ? bias[n] : 0.f;
#pragma unroll
    for (int r = 0; r < 8; r++) {
      int m = m0 + wave * 16 + half * 8 + r;
      float v = acc[ct][r] * scale;
      if (GATE) v *= gate[(size_t)m * (N >> 4) + ((n0 >> 4) + ct)];
      size_t idx = (size_t)m * N + n;
      if (OUT_F16)       ((_Float16*)Cout)[idx] = (_Float16)v;
      else if (ACC)      ((float*)Cout)[idx] += v;
      else               ((float*)Cout)[idx] = v + bval;
    }
  }
}

// ---------------------------------------------------------------------------
// RoPE + layout pack:  q,k,v f32 [B,S,H*D] ->
//   qh,kh f16 [B,H,S,D] (rotated),  vt f16 [B,H,D,S] (transposed)
// ---------------------------------------------------------------------------
__global__ __launch_bounds__(256) void rope_pack(
    const float* __restrict__ q, const float* __restrict__ k,
    const float* __restrict__ v, _Float16* __restrict__ qh,
    _Float16* __restrict__ kh, _Float16* __restrict__ vt) {
  int idx = blockIdx.x * 256 + threadIdx.x;     // over B*S*H*D
  int d = idx & (DDIM - 1);
  int h = (idx >> 7) & (HH - 1);
  int s = (idx >> 11) & (SS - 1);
  int b = idx >> 22;
  size_t tin = ((size_t)(b * SS + s)) * HD + h * DDIM + d;
  int i = d & 63;                               // freq index
  float inv = __expf(-(float)i * (9.2103403719761836f / 64.f));  // theta^-i/64
  float fr = (float)s * inv;
  float sn, cs;
  __sincosf(fr, &sn, &cs);
  float qv = q[tin], kv = k[tin], qo, ko;
  if (d < 64) {
    float q2 = q[tin + 64], k2 = k[tin + 64];
    qo = qv * cs - q2 * sn;  ko = kv * cs - k2 * sn;
  } else {
    float q2 = q[tin - 64], k2 = k[tin - 64];
    qo = qv * cs + q2 * sn;  ko = kv * cs + k2 * sn;
  }
  size_t bh = (size_t)(b * HH + h);
  size_t oq = (bh * SS + s) * DDIM + d;
  qh[oq] = (_Float16)qo;
  kh[oq] = (_Float16)ko;
  vt[(bh * DDIM + d) * SS + s] = (_Float16)v[tin];
}

// ---------------------------------------------------------------------------
// Flash attention, causal. One wave32 per 16-query tile per (b,h).
// Key tiles of 32. All K and V fragments for a tile are loaded up-front so
// global loads stay in flight across the WMMA chain and the softmax VALU
// phase; every row loop is fully unrolled (no v_movrels dynamic indexing).
// ---------------------------------------------------------------------------
__global__ __launch_bounds__(32) void attn_kernel(
    const _Float16* __restrict__ qh, const _Float16* __restrict__ kh,
    const _Float16* __restrict__ vt, float* __restrict__ o) {
  __shared__ _Float16 Plds[16 * 32];
  const int q0 = blockIdx.x * 16;
  const int bh = blockIdx.y;            // b*H + h
  const int b = bh >> 4, h = bh & 15;
  const int lane = threadIdx.x;
  const int li = lane & 15, half = lane >> 4;
  const _Float16* qB = qh + (size_t)bh * SS * DDIM;
  const _Float16* kB = kh + (size_t)bh * SS * DDIM;
  const _Float16* vB = vt + (size_t)bh * DDIM * SS;

  v16h qf[4];
#pragma unroll
  for (int dc = 0; dc < 4; dc++)
    qf[dc] = loadFragA(qB + (size_t)q0 * DDIM + dc * 32, DDIM, li, half);

  v8f zero = {0.f, 0.f, 0.f, 0.f, 0.f, 0.f, 0.f, 0.f};
  float mrow[8], lrow[8];
  v8f oacc[8];
#pragma unroll
  for (int r = 0; r < 8; r++) { mrow[r] = -INFINITY; lrow[r] = 0.f; oacc[r] = zero; }

  const int ntiles = (q0 + 16 + 31) >> 5;
  const float sc = 0.08838834764831845f;   // 1/sqrt(128)

  for (int kt = 0; kt < ntiles; kt++) {
    const int kb = kt * 32;

    // Issue ALL global loads for this tile first: 8 K-fragments + 8
    // V-fragments (V only consumed after softmax -> big overlap window).
    v16h kf0[4], kf1[4], vfr[8];
#pragma unroll
    for (int dc = 0; dc < 4; dc++) {
      kf0[dc] = loadFragB(kB + (size_t)kb * DDIM + dc * 32, DDIM, li, half);
      kf1[dc] = loadFragB(kB + (size_t)(kb + 16) * DDIM + dc * 32, DDIM, li, half);
    }
#pragma unroll
    for (int ct = 0; ct < 8; ct++)
      vfr[ct] = loadFragB(vB + (size_t)(ct * 16) * SS + kb, SS, li, half);

    v8f s0 = zero, s1 = zero;
#pragma unroll
    for (int dc = 0; dc < 4; dc++) {
      s0 = wmma16(qf[dc], kf0[dc], s0);
      s1 = wmma16(qf[dc], kf1[dc], s1);
    }

    float alpha[8];
#pragma unroll
    for (int r = 0; r < 8; r++) {
      int m = half * 8 + r, qrow = q0 + m;
      float a0 = s0[r] * sc; if (kb + li > qrow)       a0 = -1e30f;
      float a1 = s1[r] * sc; if (kb + 16 + li > qrow)  a1 = -1e30f;
      float t = fmaxf(a0, a1);
      t = fmaxf(t, __shfl_xor(t, 1, 16));
      t = fmaxf(t, __shfl_xor(t, 2, 16));
      t = fmaxf(t, __shfl_xor(t, 4, 16));
      t = fmaxf(t, __shfl_xor(t, 8, 16));
      float mn = fmaxf(mrow[r], t);
      alpha[r] = __expf(mrow[r] - mn);
      float p0 = __expf(a0 - mn), p1 = __expf(a1 - mn);
      float ps = p0 + p1;
      ps += __shfl_xor(ps, 1, 16);
      ps += __shfl_xor(ps, 2, 16);
      ps += __shfl_xor(ps, 4, 16);
      ps += __shfl_xor(ps, 8, 16);
      lrow[r] = lrow[r] * alpha[r] + ps;
      mrow[r] = mn;
      Plds[m * 32 + li]      = (_Float16)p0;
      Plds[m * 32 + 16 + li] = (_Float16)p1;
    }
#pragma unroll
    for (int ct = 0; ct < 8; ct++)
#pragma unroll
      for (int r = 0; r < 8; r++) oacc[ct][r] *= alpha[r];

    // cross-lane LDS RAW within the wave: drain DS before reading P back
    asm volatile("s_wait_dscnt 0" ::: "memory");
    v16h pf = loadFragA(Plds, 32, li, half);
#pragma unroll
    for (int ct = 0; ct < 8; ct++)
      oacc[ct] = wmma16(pf, vfr[ct], oacc[ct]);
  }

#pragma unroll
  for (int r = 0; r < 8; r++) lrow[r] = 1.f / lrow[r];
  // o written token-major [B,S,H*D] for the output projection GEMM
#pragma unroll
  for (int ct = 0; ct < 8; ct++)
#pragma unroll
    for (int r = 0; r < 8; r++) {
      int m = half * 8 + r;
      size_t idx = ((size_t)(b * SS + q0 + m)) * HD + h * DDIM + ct * 16 + li;
      o[idx] = oacc[ct][r] * lrow[r];
    }
}

// ---------------------------------------------------------------------------
// Host orchestration
// ---------------------------------------------------------------------------
extern "C" void kernel_launch(void* const* d_in, const int* in_sizes, int n_in,
                              void* d_out, int out_size, void* d_ws, size_t ws_size,
                              hipStream_t stream) {
  (void)in_sizes; (void)n_in; (void)out_size; (void)ws_size;
  const float* x    = (const float*)d_in[0];
  const float* mask = (const float*)d_in[1];   // [B,S,NEXP]
  const float* Wq = (const float*)d_in[2];  const float* bq = (const float*)d_in[3];
  const float* Wk = (const float*)d_in[4];  const float* bk = (const float*)d_in[5];
  const float* Wv = (const float*)d_in[6];  const float* bv = (const float*)d_in[7];
  const float* Wo = (const float*)d_in[8];
  const float* Aq = (const float*)d_in[9];  const float* Bq = (const float*)d_in[10];
  const float* Ak = (const float*)d_in[11]; const float* Bk = (const float*)d_in[12];
  const float* Av = (const float*)d_in[13]; const float* Bv = (const float*)d_in[14];
  const float* Ao = (const float*)d_in[15]; const float* Bo = (const float*)d_in[16];
  float* out = (float*)d_out;

  char* ws = (char*)d_ws;
  const size_t MiB = 1ull << 20;
  _Float16* xh  = (_Float16*)(ws + 0);           // 16 MiB  (reused as qh)
  _Float16* ah  = (_Float16*)(ws + 16 * MiB);    //  1 MiB
  float*    qf  = (float*)   (ws + 17 * MiB);    // 32 MiB  (reused as o f32)
  float*    kf  = (float*)   (ws + 49 * MiB);    // 32 MiB  (reused as oh f16)
  float*    vf  = (float*)   (ws + 81 * MiB);    // 32 MiB
  _Float16* khh = (_Float16*)(ws + 113 * MiB);   // 16 MiB
  _Float16* vth = (_Float16*)(ws + 129 * MiB);   // 16 MiB
  _Float16* qhh = xh;                            // alias after x no longer needed
  float*    of  = qf;                            // attention output f32
  _Float16* oh  = (_Float16*)kf;                 // attention output f16

  const int M = MTOK, E = EMBD, N = HD, NL = NLORA;
  const int nElem = M * E;                       // 8388608

  // 1) x -> f16
  cvt_f32_to_f16<<<dim3(nElem / 4 / 256), dim3(256), 0, stream>>>(x, xh, nElem / 4);

  dim3 blk(256);
  dim3 gA(NL / NT, M / MT);   // (2, 32)  LoRA-A GEMMs
  dim3 gM(N / NT, M / MT);    // (32, 32) main / LoRA-B GEMMs

  // 2) Q/K/V projections: gated LoRA-A -> dense+bias -> accumulate LoRA-B
  gemm_kernel<true,  true,  false><<<gA, blk, 0, stream>>>(xh, Aq, nullptr, mask, 1.f, ah, M, NL, E);
  gemm_kernel<false, false, false><<<gM, blk, 0, stream>>>(xh, Wq, bq, nullptr, 1.f, qf, M, N, E);
  gemm_kernel<false, false, true ><<<gM, blk, 0, stream>>>(ah, Bq, nullptr, nullptr, 1.f, qf, M, N, NL);

  gemm_kernel<true,  true,  false><<<gA, blk, 0, stream>>>(xh, Ak, nullptr, mask, 1.f, ah, M, NL, E);
  gemm_kernel<false, false, false><<<gM, blk, 0, stream>>>(xh, Wk, bk, nullptr, 1.f, kf, M, N, E);
  gemm_kernel<false, false, true ><<<gM, blk, 0, stream>>>(ah, Bk, nullptr, nullptr, 1.f, kf, M, N, NL);

  gemm_kernel<true,  true,  false><<<gA, blk, 0, stream>>>(xh, Av, nullptr, mask, 1.f, ah, M, NL, E);
  gemm_kernel<false, false, false><<<gM, blk, 0, stream>>>(xh, Wv, bv, nullptr, 1.f, vf, M, N, E);
  gemm_kernel<false, false, true ><<<gM, blk, 0, stream>>>(ah, Bv, nullptr, nullptr, 1.f, vf, M, N, NL);

  // 3) RoPE + pack to attention layouts (overwrites xh with qh)
  rope_pack<<<dim3(nElem / 256), dim3(256), 0, stream>>>(qf, kf, vf, qhh, khh, vth);

  // 4) Causal flash attention -> o f32 [B,S,H*D]
  attn_kernel<<<dim3(SS / 16, BB * HH), dim3(32), 0, stream>>>(qhh, khh, vth, of);

  // 5) o -> f16
  cvt_f32_to_f16<<<dim3(nElem / 4 / 256), dim3(256), 0, stream>>>(of, oh, nElem / 4);

  // 6) Output projection (no bias) + LoRA, straight into d_out
  gemm_kernel<true,  true,  false><<<gA, blk, 0, stream>>>(oh, Ao, nullptr, mask, 1.f, ah, M, NL, N);
  gemm_kernel<false, false, false><<<gM, blk, 0, stream>>>(oh, Wo, nullptr, nullptr, 1.f, out, M, E, N);
  gemm_kernel<false, false, true ><<<gM, blk, 0, stream>>>(ah, Bo, nullptr, nullptr, 1.f, out, M, E, NL);
}
